// QMDPLayer_22952305230020
// MI455X (gfx1250) — compile-verified, matrix operand order checked by reading, not run on hardware
//
#include <hip/hip_runtime.h>
#include <math.h>
#include <stdint.h>

// Problem dims (fixed by the reference)
#define S_DIM 256
#define A_DIM 32
#define R_DIM 64
#define H_DIM 30
#define T_LEN 32
#define B_DIM 128
#define K_DIM (A_DIM * S_DIM) // 8192 contraction length of belief GEMM

typedef __attribute__((ext_vector_type(16))) __bf16    v16bf;
typedef __attribute__((ext_vector_type(8)))  float     v8f;
typedef __attribute__((ext_vector_type(8)))  uint32_t  v8u;
typedef __attribute__((ext_vector_type(4)))  uint32_t  v4u;

__device__ __forceinline__ uint16_t f32_to_bf16_rne(float f) {
  uint32_t u = __builtin_bit_cast(uint32_t, f);
  uint32_t r = 0x7FFFu + ((u >> 16) & 1u);
  return (uint16_t)((u + r) >> 16);
}

__device__ __forceinline__ float wave_max(float v) {
  #pragma unroll
  for (int off = 16; off > 0; off >>= 1) v = fmaxf(v, __shfl_xor(v, off, 32));
  return v;
}
__device__ __forceinline__ float wave_sum(float v) {
  #pragma unroll
  for (int off = 16; off > 0; off >>= 1) v += __shfl_xor(v, off, 32);
  return v;
}

// ---------------- Precompute (once per launch) ----------------

// G[i,j] = u[i,:] . u[j,:]   (256x256, tiny)
__global__ void k_gram(const float* __restrict__ u, float* __restrict__ G) {
  int i = blockIdx.x;      // 256
  int j = threadIdx.x;     // 256
  const float* ui = u + i * R_DIM;
  const float* uj = u + j * R_DIM;
  float acc = 0.f;
  #pragma unroll
  for (int m = 0; m < R_DIM; ++m) acc += ui[m] * uj[m];
  G[i * S_DIM + j] = acc;
}

// p[k,i] = (u[i,:] . v[k,:]) / ||v[k,:]||
__global__ void k_proj(const float* __restrict__ u, const float* __restrict__ v,
                       float* __restrict__ p) {
  int k = blockIdx.x;      // 32
  int i = threadIdx.x;     // 256
  __shared__ float vsh[R_DIM];
  __shared__ float inv_norm;
  if (i < R_DIM) vsh[i] = v[k * R_DIM + i];
  __syncthreads();
  if (i == 0) {
    float s = 0.f;
    #pragma unroll
    for (int m = 0; m < R_DIM; ++m) s += vsh[m] * vsh[m];
    inv_norm = rsqrtf(fmaxf(s, 1e-30f));
  }
  __syncthreads();
  const float* ui = u + i * R_DIM;
  float acc = 0.f;
  #pragma unroll
  for (int m = 0; m < R_DIM; ++m) acc += ui[m] * vsh[m];
  p[k * S_DIM + i] = acc * inv_norm;
}

// transition[k,i,j] = softmax_j( G[i,j] - 2 p[k,i] p[k,j] ), stored TRANSPOSED
// as bf16: Tt[j][k*256+i]  (row j contiguous over ki -> WMMA B-fragment rows)
__global__ void k_transition(const float* __restrict__ G, const float* __restrict__ p,
                             uint16_t* __restrict__ Tt) {
  int wid  = (blockIdx.x * blockDim.x + threadIdx.x) >> 5; // 0..8191 (one wave per (k,i))
  int lane = threadIdx.x & 31;
  int k = wid >> 8, i = wid & 255;
  float pki = p[k * S_DIM + i];
  float w[8];
  float mx = -3.4e38f;
  #pragma unroll
  for (int c = 0; c < 8; ++c) {
    int j = lane + 32 * c;
    float val = G[i * S_DIM + j] - 2.f * pki * p[k * S_DIM + j];
    w[c] = val;
    mx = fmaxf(mx, val);
  }
  mx = wave_max(mx);
  float s = 0.f;
  #pragma unroll
  for (int c = 0; c < 8; ++c) { w[c] = __expf(w[c] - mx); s += w[c]; }
  s = wave_sum(s);
  float inv = 1.f / s;
  #pragma unroll
  for (int c = 0; c < 8; ++c) {
    int j = lane + 32 * c;
    Tt[(size_t)j * K_DIM + (k * S_DIM + i)] = f32_to_bf16_rne(w[c] * inv);
  }
}

// truncated-poisson horizon weights (identical for all n): tau_pdf[30]
__global__ void k_taupdf(const float* __restrict__ tau, float* __restrict__ tau_pdf) {
  int h = threadIdx.x; // 32 lanes
  float lt = fminf(fmaxf(tau[0], logf(1e-6f)), logf(1000.f));
  float rate = __expf(lt);
  float lp = -3.4e38f;
  if (h < H_DIM) {
    float ks = (float)(h + 1);
    lp = ks * lt - rate - lgammaf(ks + 1.f);
  }
  float mx = wave_max(lp);
  float e = (h < H_DIM) ? __expf(lp - mx) : 0.f;
  float s = wave_sum(e);
  if (h < H_DIM) tau_pdf[h] = e / s;
}

// alpha_a = softmax(logp_u, axis=-1), one wave per (t,n) row of 32
__global__ void k_alpha_a(const float* __restrict__ logp_u, float* __restrict__ alpha_a) {
  int row  = (blockIdx.x * blockDim.x + threadIdx.x) >> 5; // t*B + n
  int k    = threadIdx.x & 31;
  float x  = logp_u[row * A_DIM + k];
  float mx = wave_max(x);
  float e  = __expf(x - mx);
  float s  = wave_sum(e);
  alpha_a[row * A_DIM + k] = e / s;
}

__global__ void k_copy(const float* __restrict__ src, float* __restrict__ dst, int n) {
  int i = blockIdx.x * blockDim.x + threadIdx.x;
  if (i < n) dst[i] = src[i];
}

// ---------------- Per-step pipeline ----------------

// c[n, k*256+i] = alpha_a[t,n,k] * b_cur[n,i]  (bf16, WMMA A-operand rows)
__global__ void k_prep_c(const float* __restrict__ alpha_a_t, const float* __restrict__ b_cur,
                         uint16_t* __restrict__ c) {
  int idx = blockIdx.x * blockDim.x + threadIdx.x; // 128*8192
  int n  = idx >> 13;
  int ki = idx & (K_DIM - 1);
  int k  = ki >> 8;
  int i  = ki & 255;
  c[idx] = f32_to_bf16_rne(alpha_a_t[n * A_DIM + k] * b_cur[n * S_DIM + i]);
}

// s_next = c @ Tt^T : [128 x 8192] x [8192 x 256], bf16 in / f32 accumulate.
// One wave per 16x16 output tile; 8 waves per block; 16 blocks -> 128 tiles.
__global__ void __launch_bounds__(256)
k_belief_gemm(const uint16_t* __restrict__ c, const uint16_t* __restrict__ Tt,
              float* __restrict__ s_next) {
  int tile = blockIdx.x * 8 + (threadIdx.x >> 5); // 0..127
  int mt = tile >> 4;   // 0..7  (rows of 16 over B=128)
  int nt = tile & 15;   // 0..15 (cols of 16 over S=256)
  int lane = threadIdx.x & 31;
  int lm   = lane & 15;
  int half = lane >> 4;

  const uint32_t* arow = (const uint32_t*)(c  + (size_t)(mt * 16 + lm) * K_DIM);
  const uint32_t* brow = (const uint32_t*)(Tt + (size_t)(nt * 16 + lm) * K_DIM);

  v8f acc = {};
  #pragma unroll 4
  for (int kk = 0; kk < K_DIM; kk += 32) {
    // 16-bit A/B fragment: VGPR p holds K-pair (p/4)*16 + half*8 + (p%4)*2
    // -> two contiguous 16-byte loads per operand per lane.
    int d0 = (kk + half * 8) >> 1;        // dword index of K-pairs 0..3
    int d1 = (kk + 16 + half * 8) >> 1;   // dword index of K-pairs 4..7
    v8u aw, bw;
    aw.lo = *(const v4u*)(arow + d0);
    aw.hi = *(const v4u*)(arow + d1);
    bw.lo = *(const v4u*)(brow + d0);
    bw.hi = *(const v4u*)(brow + d1);
    v16bf A = __builtin_bit_cast(v16bf, aw);
    v16bf B = __builtin_bit_cast(v16bf, bw);
    acc = __builtin_amdgcn_wmma_f32_16x16x32_bf16(false, A, false, B,
                                                  (short)0, acc, false, false);
  }
  // D layout: VGPR r, lanes0-15 -> M=r, lanes16-31 -> M=8+r; N=lane%16
  #pragma unroll
  for (int r = 0; r < 8; ++r) {
    int m = mt * 16 + half * 8 + r;
    s_next[(size_t)m * S_DIM + nt * 16 + lm] = acc[r];
  }
}

// b_post = softmax_j( log(s_next+eps) + logp_o[t] ); write to b_cur and alpha_b[t]
__global__ void k_belief_softmax(const float* __restrict__ s_next,
                                 const float* __restrict__ logp_o_t,
                                 float* __restrict__ b_cur,
                                 float* __restrict__ alpha_b_t) {
  int n    = (blockIdx.x * blockDim.x + threadIdx.x) >> 5; // row, one wave each
  int lane = threadIdx.x & 31;
  float x[8];
  float mx = -3.4e38f;
  #pragma unroll
  for (int cidx = 0; cidx < 8; ++cidx) {
    int j = lane + 32 * cidx;
    float val = __logf(s_next[n * S_DIM + j] + 1e-6f) + logp_o_t[n * S_DIM + j];
    x[cidx] = val;
    mx = fmaxf(mx, val);
  }
  mx = wave_max(mx);
  float s = 0.f;
  #pragma unroll
  for (int cidx = 0; cidx < 8; ++cidx) { x[cidx] = __expf(x[cidx] - mx); s += x[cidx]; }
  s = wave_sum(s);
  float inv = 1.f / s;
  #pragma unroll
  for (int cidx = 0; cidx < 8; ++cidx) {
    int j = lane + 32 * cidx;
    float out = x[cidx] * inv;
    b_cur[n * S_DIM + j] = out;
    alpha_b_t[n * S_DIM + j] = out;
  }
}

// pi[n,k] = sum_h tau_pdf[h] * softmax_k( sum_i b[n,i] value[h,n,k,i] )
// Batched matvec (matrix differs per (h,n)) -> L2-bandwidth bound, VALU path.
__global__ void __launch_bounds__(256)
k_policy(const float* __restrict__ value, const float* __restrict__ b_cur,
         const float* __restrict__ tau_pdf, float* __restrict__ alpha_pi_t) {
  int n    = blockIdx.x;       // 128
  int tid  = threadIdx.x;      // 256
  int wave = tid >> 5;
  int lane = tid & 31;
  __shared__ float bsh[S_DIM];
  __shared__ float logits[A_DIM];
  __shared__ float accsh[A_DIM];
  bsh[tid] = b_cur[n * S_DIM + tid];
  if (tid < A_DIM) accsh[tid] = 0.f;
  __syncthreads();
  for (int h = 0; h < H_DIM; ++h) {
    const float* vh = value + ((size_t)h * B_DIM + n) * (A_DIM * S_DIM);
    #pragma unroll
    for (int kc = 0; kc < 4; ++kc) {
      int k = wave + kc * 8;
      const float4* vp = (const float4*)(vh + k * S_DIM) + lane;
      float4 va = vp[0];    // i = lane*4 .. +3
      float4 vb = vp[32];   // i = 128 + lane*4 .. +3
      int i0 = lane * 4;
      float d = va.x * bsh[i0]       + va.y * bsh[i0 + 1]
              + va.z * bsh[i0 + 2]   + va.w * bsh[i0 + 3]
              + vb.x * bsh[128 + i0] + vb.y * bsh[128 + i0 + 1]
              + vb.z * bsh[128 + i0 + 2] + vb.w * bsh[128 + i0 + 3];
      d = wave_sum(d);
      if (lane == 0) logits[k] = d;
    }
    __syncthreads();
    if (wave == 0) {
      float x  = logits[lane];          // 32 lanes == A_DIM
      float mx = wave_max(x);
      float e  = __expf(x - mx);
      float s  = wave_sum(e);
      accsh[lane] += tau_pdf[h] * (e / s);
    }
    __syncthreads();
  }
  if (tid < A_DIM) alpha_pi_t[n * A_DIM + tid] = accsh[tid];
}

// ---------------- Host-side orchestration ----------------

extern "C" void kernel_launch(void* const* d_in, const int* in_sizes, int n_in,
                              void* d_out, int out_size, void* d_ws, size_t ws_size,
                              hipStream_t stream) {
  (void)in_sizes; (void)n_in; (void)out_size; (void)ws_size;
  const float* logp_o = (const float*)d_in[0]; // [T,B,S]
  const float* logp_u = (const float*)d_in[1]; // [T,B,A]
  const float* value  = (const float*)d_in[2]; // [H,B,A,S]
  const float* b_in   = (const float*)d_in[3]; // [B,S]
  const float* tau    = (const float*)d_in[4]; // [1,1]
  const float* u      = (const float*)d_in[5]; // [S,R]
  const float* v      = (const float*)d_in[6]; // [A,R]

  float* alpha_b  = (float*)d_out;                                    // [T,B,S]
  float* alpha_pi = (float*)d_out + (size_t)T_LEN * B_DIM * S_DIM;    // [T,B,A]

  char* ws = (char*)d_ws;
  size_t off = 0;
  auto wsalloc = [&](size_t bytes) -> void* {
    off = (off + 255) & ~(size_t)255;
    void* p = ws + off;
    off += bytes;
    return p;
  };
  uint16_t* Tt      = (uint16_t*)wsalloc((size_t)S_DIM * K_DIM * 2);   // 4 MB
  float*    G       = (float*)   wsalloc((size_t)S_DIM * S_DIM * 4);
  float*    p_ws    = (float*)   wsalloc((size_t)A_DIM * S_DIM * 4);
  float*    taupdf  = (float*)   wsalloc(32 * 4);
  float*    alpha_a = (float*)   wsalloc((size_t)T_LEN * B_DIM * A_DIM * 4);
  float*    b_cur   = (float*)   wsalloc((size_t)B_DIM * S_DIM * 4);
  uint16_t* c_mat   = (uint16_t*)wsalloc((size_t)B_DIM * K_DIM * 2);   // 2 MB
  float*    s_next  = (float*)   wsalloc((size_t)B_DIM * S_DIM * 4);

  // One-time precompute
  k_gram      <<<S_DIM, S_DIM, 0, stream>>>(u, G);
  k_proj      <<<A_DIM, S_DIM, 0, stream>>>(u, v, p_ws);
  k_transition<<<(A_DIM * S_DIM * 32) / 256, 256, 0, stream>>>(G, p_ws, Tt);
  k_taupdf    <<<1, 32, 0, stream>>>(tau, taupdf);
  k_alpha_a   <<<(T_LEN * B_DIM * 32) / 256, 256, 0, stream>>>(logp_u, alpha_a);
  k_copy      <<<(B_DIM * S_DIM + 255) / 256, 256, 0, stream>>>(b_in, b_cur, B_DIM * S_DIM);

  // Sequential scan over T (hard dependence through b_cur)
  for (int t = 0; t < T_LEN; ++t) {
    k_prep_c        <<<(B_DIM * K_DIM) / 256, 256, 0, stream>>>(
        alpha_a + (size_t)t * B_DIM * A_DIM, b_cur, c_mat);
    k_belief_gemm   <<<16, 256, 0, stream>>>(c_mat, Tt, s_next);
    k_belief_softmax<<<B_DIM / 8, 256, 0, stream>>>(
        s_next, logp_o + (size_t)t * B_DIM * S_DIM, b_cur,
        alpha_b + (size_t)t * B_DIM * S_DIM);
    k_policy        <<<B_DIM, 256, 0, stream>>>(
        value, b_cur, taupdf, alpha_pi + (size_t)t * B_DIM * A_DIM);
  }
}
